// FragExtraStackIteration_62663572848704
// MI455X (gfx1250) — matrix operand order repaired and checked by reading, not compiled, not run on hardware
//
#include <hip/hip_runtime.h>
#include <cstring>

typedef unsigned short u16;
typedef unsigned int u32;

// ---------------------------------------------------------------------------
// problem dims
// ---------------------------------------------------------------------------
#define R_    256
#define A_    48
#define M_    8
#define NTOT  304     // R + A
#define NPAD  320     // padded softmax / V length (multiple of 32)
#define C1_   64
#define C2_   128
#define AC_   32
#define H_    8
#define MID_  32

// ---------------------------------------------------------------------------
// bf16 helpers + WMMA fragment plumbing (CDNA5 wave32, 16x16x32 bf16)
// ---------------------------------------------------------------------------
typedef __attribute__((ext_vector_type(16))) __bf16 v16bf;
typedef __attribute__((ext_vector_type(8)))  float  v8f;

union FragAB { v16bf v; uint4 u[2]; };

__device__ __forceinline__ u16 f2bf(float f) {
  u32 u = __float_as_uint(f);
  u32 r = u + 0x7FFFu + ((u >> 16) & 1u);     // round-to-nearest-even
  return (u16)(r >> 16);
}
__device__ __forceinline__ float bf2f(u16 h) { return __uint_as_float(((u32)h) << 16); }

// A matrix (16xK tile, row-major, k contiguous): lane holds row = lane&15.
// ISA layout: lanes 0-15 K=[0..8)+[16..24), lanes 16-31 K=[8..16)+[24..32)
__device__ __forceinline__ v16bf load_a_frag(const u16* Abase, int lda, int k0, int lane) {
  int r = lane & 15, half = lane >> 4;
  const u16* p = Abase + (long long)r * lda + k0 + half * 8;
  FragAB f;
  f.u[0] = *(const uint4*)(p);
  f.u[1] = *(const uint4*)(p + 16);
  return f.v;
}
// B matrix stored transposed [N x K] row-major: lane holds col n = lane&15.
// ISA layout: lanes 0-15 K=[0..8)+[16..24), lanes 16-31 K=[8..16)+[24..32)
__device__ __forceinline__ v16bf load_b_frag(const u16* Bbase, int ldb, int k0, int lane) {
  int r = lane & 15, half = lane >> 4;
  const u16* p = Bbase + (long long)r * ldb + k0 + half * 8;
  FragAB f;
  f.u[0] = *(const uint4*)(p);
  f.u[1] = *(const uint4*)(p + 16);
  return f.v;
}
__device__ __forceinline__ v8f wmma_bf16(v16bf a, v16bf b, v8f c) {
  return __builtin_amdgcn_wmma_f32_16x16x32_bf16(false, a, false, b, (short)0, c, false, false);
}

// ---------------------------------------------------------------------------
// generic dual-batched WMMA GEMM:  C[b1,b2] = act(scale*A·Bᵀ + bias + add2d + resid)
// one wave per 16x16 output tile; all dims multiples of 16 (K of 32)
// ---------------------------------------------------------------------------
struct GArgs {
  const u16* A;  int lda;  long long aB1, aB2;
  const u16* BT; int ldb;  long long bB1, bB2;
  float* outF;   long long oRow, oB1, oB2;
  u16*   outH;   long long hRow, hB1, hB2;
  const float* bias;                                     // per-column
  const float* add2d; long long addRow, addB1, addB2; int addCol;
  const float* resid; long long resRow, resB1, resB2;
  int M, N, K;
  int nb1;
  int relu, accumulate, copies; long long copyStride;
  float scale;
};

__global__ __launch_bounds__(32) void k_gemm(GArgs g) {
  int lane = threadIdx.x;
  int mt = blockIdx.x, nt = blockIdx.y;
  int b1 = blockIdx.z % g.nb1, b2 = blockIdx.z / g.nb1;
  const u16* Abase = g.A  + g.aB1 * b1 + g.aB2 * b2 + (long long)(mt * 16) * g.lda;
  const u16* Bbase = g.BT + g.bB1 * b1 + g.bB2 * b2 + (long long)(nt * 16) * g.ldb;
  v8f acc = {};
  for (int k0 = 0; k0 < g.K; k0 += 32) {
    v16bf a = load_a_frag(Abase, g.lda, k0, lane);
    v16bf b = load_b_frag(Bbase, g.ldb, k0, lane);
    acc = wmma_bf16(a, b, acc);
  }
  int half = lane >> 4, n = lane & 15;
  int col = nt * 16 + n;
  for (int r = 0; r < 8; r++) {
    int row = mt * 16 + r + 8 * half;
    float v = acc[r] * g.scale;
    if (g.bias)  v += g.bias[col];
    if (g.add2d) v += g.add2d[g.addB1 * b1 + g.addB2 * b2 + (long long)row * g.addRow + (long long)col * g.addCol];
    if (g.resid) v += g.resid[g.resB1 * b1 + g.resB2 * b2 + (long long)row * g.resRow + col];
    if (g.relu)  v = v > 0.f ? v : 0.f;
    if (g.outF) {
      long long o = g.oB1 * b1 + g.oB2 * b2 + (long long)row * g.oRow + col;
      float base = g.accumulate ? g.outF[o] : 0.f;
      float w = base + v;
      for (int cp = 0; cp < g.copies; cp++) g.outF[o + (long long)cp * g.copyStride] = w;
    }
    if (g.outH)
      g.outH[g.hB1 * b1 + g.hB2 * b2 + (long long)row * g.hRow + col] = f2bf(v);
  }
}

// ---------------------------------------------------------------------------
// small utility kernels
// ---------------------------------------------------------------------------
__global__ void k_ln(const float* x, int rows, int C, const float* gw, const float* bw, u16* out) {
  int r = blockIdx.x * blockDim.x + threadIdx.x;
  if (r >= rows) return;
  const float* p = x + (long long)r * C;
  float m = 0.f;
  for (int c = 0; c < C; c++) m += p[c];
  m /= C;
  float v = 0.f;
  for (int c = 0; c < C; c++) { float d = p[c] - m; v += d * d; }
  float rs = rsqrtf(v / C + 1e-5f);
  u16* o = out + (long long)r * C;
  for (int c = 0; c < C; c++) o[c] = f2bf((p[c] - m) * rs * gw[c] + bw[c]);
}

// pack f32 weight [K x N] -> bf16 transposed [N x K]
__global__ void k_pack_wt(const float* W, int K, int N, float scale, u16* out) {
  long long i = (long long)blockIdx.x * 256 + threadIdx.x;
  if (i >= (long long)K * N) return;
  int n = (int)(i / K), k = (int)(i % K);
  out[i] = f2bf(W[(long long)k * N + n] * scale);
}

// combine 4 pair-bias projections [128x8] into packed [32 x 128], scaled by 1/sqrt(AC)
__global__ void k_pack_bias4(const float* rr, const float* rl, const float* lr, const float* ll, u16* out) {
  int i = blockIdx.x * 256 + threadIdx.x;
  if (i >= 32 * 128) return;
  int k = i & 127, n = i >> 7;
  int sec = n >> 3, h = n & 7;
  const float* p = sec == 0 ? rr : sec == 1 ? rl : sec == 2 ? lr : ll;
  out[i] = f2bf(p[k * 8 + h] * 0.17677669529663687f);
}

// pack OPM weight [1024 x 128] -> bf16 [(c*32+u) x 32] with k = v
// swap=0: u=x, v=y (contract over y);  swap=1: u=y, v=x (contract over x)
__global__ void k_pack_opm(const float* W, int swap, u16* out) {
  int i = blockIdx.x * 256 + threadIdx.x;
  if (i >= 1024 * 128) return;
  int v = i & 31, t = i >> 5;
  int u = t & 31, c = t >> 5;
  int x = swap ? v : u, y = swap ? u : v;
  out[i] = f2bf(W[(x * 32 + y) * 128 + c]);
}

__global__ void k_f2bf(const float* x, long long n, u16* out) {
  long long i = (long long)blockIdx.x * 256 + threadIdx.x;
  if (i < n) out[i] = f2bf(x[i]);
}

// split rec q6 [256 x 1536] into per-head tensors (sections: lq lk lv rq rk rv)
__global__ void k_repack_rec(const u16* q6, u16* rq, u16* rk, u16* lq, u16* lk,
                             u16* rvT, u16* vcomb) {
  int i = blockIdx.x * 256 + threadIdx.x;
  if (i >= 8 * 256 * 32) return;
  int c = i & 31, t = i >> 5;
  int row = t & 255, h = t >> 8;
  const u16* src = q6 + (long long)row * 1536 + c * 48 + h;
  int o = (h * 256 + row) * 32 + c;
  lq[o] = src[0];  lk[o] = src[8];
  rq[o] = src[24]; rk[o] = src[32];
  rvT[(h * 32 + c) * 256 + row] = src[40];                 // rec_rv^T [h][c][r]
  u16 lv = src[16];                                        // rec_lv -> Vcomb k<256
  for (int m = 0; m < 8; m++)
    vcomb[((m * 8 + h) * 32 + c) * NPAD + row] = lv;
}

// split lig q6 [384 x 1536] into per-head tensors (sections: lq lk lv rq rk rv)
__global__ void k_repack_lig(const u16* q6, u16* Lrk, u16* Lrq, u16* Llq, u16* Llk,
                             u16* LrvT, u16* vcomb) {
  int i = blockIdx.x * 256 + threadIdx.x;
  if (i >= 8 * 384 * 32) return;
  int c = i & 31, t = i >> 5;
  int row = t % 384, h = t / 384;
  int m = row / 48, a = row % 48;
  const u16* src = q6 + (long long)row * 1536 + c * 48 + h;
  int o = (h * 384 + row) * 32 + c;
  Llq[o] = src[0];  Llk[o] = src[8];
  Lrq[o] = src[24]; Lrk[o] = src[32];
  float rv = bf2f(src[40]) * 0.125f;                       // lig_rv scaled by 1/M
  LrvT[((h * 8 + m) * 32 + c) * 64 + a] = f2bf(rv);
  if (a < 16) LrvT[((h * 8 + m) * 32 + c) * 64 + 48 + a] = 0;
  vcomb[((m * 8 + h) * 32 + c) * NPAD + 256 + a] = src[16]; // lig_lv -> Vcomb
  if (a < 16) vcomb[((m * 8 + h) * 32 + c) * NPAD + 304 + a] = 0;
}

// row softmax over j (length 304, padded to 320 with zeros), one wave per row
__global__ __launch_bounds__(32) void k_softmax(const float* logits, u16* probs) {
  long long row = blockIdx.x;
  const float* p = logits + row * NPAD;
  u16* o = probs + row * NPAD;
  int lane = threadIdx.x;
  float vals[10];
  float mx = -1e30f;
  for (int t = 0; t < 10; t++) {
    int j = lane + 32 * t;
    float v = (j < NTOT) ? p[j] : -1e30f;
    vals[t] = v;
    mx = fmaxf(mx, v);
  }
  for (int s = 16; s > 0; s >>= 1) mx = fmaxf(mx, __shfl_xor(mx, s, 32));
  float sum = 0.f;
  for (int t = 0; t < 10; t++) {
    float e = (vals[t] > -1e29f) ? __expf(vals[t] - mx) : 0.f;
    vals[t] = e; sum += e;
  }
  for (int s = 16; s > 0; s >>= 1) sum += __shfl_xor(sum, s, 32);
  float inv = 1.f / sum;
  for (int t = 0; t < 10; t++) o[lane + 32 * t] = f2bf(vals[t] * inv);
}

// rec attention output: probs[0,h,i,:R]·rec_rv  +  mean_m probs[m,h,i,R:]·lig_rv
__global__ __launch_bounds__(32) void k_att_rec(const u16* probs, const u16* rvT,
                                                const u16* LrvT, u16* att) {
  int lane = threadIdx.x;
  int it = blockIdx.x, h = blockIdx.y;
  v8f acc0 = {}, acc1 = {};
  const u16* Ab = probs + (long long)(h * NTOT + it * 16) * NPAD;    // m = 0
  const u16* B0 = rvT + h * 32 * 256;
  for (int k0 = 0; k0 < 256; k0 += 32) {
    v16bf a = load_a_frag(Ab, NPAD, k0, lane);
    acc0 = wmma_bf16(a, load_b_frag(B0, 256, k0, lane), acc0);
    acc1 = wmma_bf16(a, load_b_frag(B0 + 16 * 256, 256, k0, lane), acc1);
  }
  for (int m = 0; m < 8; m++) {
    const u16* Am = probs + (long long)((m * 8 + h) * NTOT + it * 16) * NPAD + 256;
    const u16* Bm = LrvT + (long long)((h * 8 + m) * 32) * 64;       // pre-scaled 1/M
    for (int k0 = 0; k0 < 64; k0 += 32) {
      v16bf a = load_a_frag(Am, NPAD, k0, lane);
      acc0 = wmma_bf16(a, load_b_frag(Bm, 64, k0, lane), acc0);
      acc1 = wmma_bf16(a, load_b_frag(Bm + 16 * 64, 64, k0, lane), acc1);
    }
  }
  int half = lane >> 4, n = lane & 15;
  for (int r = 0; r < 8; r++) {
    int row = it * 16 + r + 8 * half;
    att[row * 256 + n * 8 + h]        = f2bf(acc0[r]);
    att[row * 256 + (16 + n) * 8 + h] = f2bf(acc1[r]);
  }
}

// lig attention output: probs[m,h,R+a,:]·Vcomb (rec_lv | lig_lv)
__global__ __launch_bounds__(32) void k_att_lig(const u16* probs, const u16* vcomb, u16* att) {
  int lane = threadIdx.x;
  int at = blockIdx.x, m = blockIdx.y, h = blockIdx.z;
  const u16* Ab = probs + (long long)((m * 8 + h) * NTOT + 256 + at * 16) * NPAD;
  const u16* Bb = vcomb + (long long)((m * 8 + h) * 32) * NPAD;
  v8f acc0 = {}, acc1 = {};
  for (int k0 = 0; k0 < NPAD; k0 += 32) {
    v16bf a = load_a_frag(Ab, NPAD, k0, lane);
    acc0 = wmma_bf16(a, load_b_frag(Bb, NPAD, k0, lane), acc0);
    acc1 = wmma_bf16(a, load_b_frag(Bb + 16 * NPAD, NPAD, k0, lane), acc1);
  }
  int half = lane >> 4, n = lane & 15;
  for (int r = 0; r < 8; r++) {
    int aa = at * 16 + r + 8 * half;
    int row = m * 48 + aa;
    att[row * 256 + n * 8 + h]        = f2bf(acc0[r]);
    att[row * 256 + (16 + n) * 8 + h] = f2bf(acc1[r]);
  }
}

// extra-column global attention core (tiny): one thread per (a, h)
__global__ void k_colattn(const float* kqv, const float* gate, u16* out) {
  int idx = blockIdx.x * 64 + threadIdx.x;
  if (idx >= 384) return;
  int a = idx >> 3, h = idx & 7;
  float q[32];
  for (int c = 0; c < 32; c++) {
    float s = 0.f;
    for (int m = 0; m < 8; m++) s += kqv[(m * 48 + a) * NPAD + c * 10 + h];
    q[c] = s * 0.125f;
  }
  float lg[8]; float mx = -1e30f;
  for (int m = 0; m < 8; m++) {
    float s = 0.f;
    for (int c = 0; c < 32; c++) s += q[c] * kqv[(m * 48 + a) * NPAD + c * 10 + 8];
    s *= 0.17677669529663687f;
    lg[m] = s; mx = fmaxf(mx, s);
  }
  float sum = 0.f;
  for (int m = 0; m < 8; m++) { lg[m] = __expf(lg[m] - mx); sum += lg[m]; }
  float inv = 1.f / sum;
  float av[32];
  for (int c = 0; c < 32; c++) {
    float s = 0.f;
    for (int m = 0; m < 8; m++) s += lg[m] * inv * kqv[(m * 48 + a) * NPAD + c * 10 + 9];
    av[c] = s;
  }
  for (int m = 0; m < 8; m++)
    for (int c = 0; c < 32; c++)
      out[(m * 48 + a) * 256 + c * 8 + h] = f2bf(gate[(m * 48 + a) * 256 + c * 8 + h] * av[c]);
}

// fused transition: x += W2·relu(W1·LN(x)+b1)+b2 ; 16 rows/block, hidden in LDS
__global__ __launch_bounds__(256) void k_transition(float* x, int C, int HD,
    const float* gw, const float* bw, const u16* W1, const float* b1,
    const u16* W2, const float* b2) {
  __shared__ float xf[16 * 128];
  __shared__ u16   xb[16 * 128];
  __shared__ u16   hb[16 * 512];
  int row0 = blockIdx.x * 16;
  int tid = threadIdx.x;
  if (tid < 16) {
    const float* p = x + (long long)(row0 + tid) * C;
    float m = 0.f;
    for (int c = 0; c < C; c++) { float v = p[c]; xf[tid * 128 + c] = v; m += v; }
    m /= C;
    float var = 0.f;
    for (int c = 0; c < C; c++) { float d = xf[tid * 128 + c] - m; var += d * d; }
    float rs = rsqrtf(var / C + 1e-5f);
    for (int c = 0; c < C; c++)
      xb[tid * 128 + c] = f2bf((xf[tid * 128 + c] - m) * rs * gw[c] + bw[c]);
  }
  __syncthreads();
  int wave = tid >> 5, lane = tid & 31;
  for (int nt = wave; nt < HD / 16; nt += 8) {
    v8f acc = {};
    const u16* Wb = W1 + (long long)nt * 16 * C;
    for (int k0 = 0; k0 < C; k0 += 32)
      acc = wmma_bf16(load_a_frag((const u16*)xb, 128, k0, lane),
                      load_b_frag(Wb, C, k0, lane), acc);
    int half = lane >> 4, n = lane & 15;
    for (int r = 0; r < 8; r++) {
      int mr = r + 8 * half, col = nt * 16 + n;
      float v = acc[r] + b1[col];
      hb[mr * 512 + col] = f2bf(v > 0.f ? v : 0.f);
    }
  }
  __syncthreads();
  for (int nt = wave; nt < C / 16; nt += 8) {
    v8f acc = {};
    const u16* Wb = W2 + (long long)nt * 16 * HD;
    for (int k0 = 0; k0 < HD; k0 += 32)
      acc = wmma_bf16(load_a_frag((const u16*)hb, 512, k0, lane),
                      load_b_frag(Wb, HD, k0, lane), acc);
    int half = lane >> 4, n = lane & 15;
    for (int r = 0; r < 8; r++) {
      int mr = r + 8 * half, col = nt * 16 + n;
      x[(long long)(row0 + mr) * C + col] = xf[mr * 128 + col] + acc[r] + b2[col];
    }
  }
}

// OPM factor extraction (rp: [256x128] = [i][x*4+sel], lp: [384x128])
__global__ void k_opm_factors(const float* rp, const float* lp,
    u16* r_ri, u16* r_rj, u16* r_li, u16* r_lj,
    u16* lm_rj, u16* lm_ri, u16* l_lj, u16* ali) {
  int t = blockIdx.x * 256 + threadIdx.x;
  if (t < 32768) {                                  // rec factors
    int sel = t / 8192, rem = t % 8192;
    int row = rem >> 5, x = rem & 31;
    u16 v = f2bf(rp[row * 128 + x * 4 + sel]);
    (sel == 0 ? r_ri : sel == 1 ? r_rj : sel == 2 ? r_li : r_lj)[row * 32 + x] = v;
    return;
  }
  t -= 32768;
  if (t < 3072) {                                   // ligand means over m
    int sel = t / 1536, rem = t % 1536;
    int ja = rem >> 5, y = rem & 31;
    float s = 0.f;
    for (int m = 0; m < 8; m++) s += lp[(m * 48 + ja) * 128 + y * 4 + (sel == 0 ? 1 : 2)];
    (sel == 0 ? lm_rj : lm_ri)[ja * 32 + y] = f2bf(s * 0.125f);
    return;
  }
  t -= 3072;
  if (t < 12288) {                                  // l_lj [m*48+jb][y]
    int row = t >> 5, y = t & 31;
    l_lj[row * 32 + y] = f2bf(lp[row * 128 + y * 4 + 3]);
    return;
  }
  t -= 12288;
  if (t < 12288) {                                  // ali [ja][m*32+x] = l_li/M
    int ja = t >> 8, k = t & 255;
    int m = k >> 5, x = k & 31;
    ali[ja * 256 + k] = f2bf(0.125f * lp[(m * 48 + ja) * 128 + x * 4 + 2]);
  }
}

// U4raw [m*48+jb][c*32+x] -> U4 [jb][c][m*32+x]
__global__ void k_repack_u4(const u16* raw, u16* out) {
  long long i = (long long)blockIdx.x * 256 + threadIdx.x;
  if (i >= 384LL * 4096) return;
  int col = (int)(i & 4095);
  int row = (int)(i >> 12);
  int m = row / 48, jb = row % 48;
  int x = col & 31, c = col >> 5;
  out[((long long)jb * 128 + c) * 256 + m * 32 + x] = raw[i];
}

// ---------------------------------------------------------------------------
// host side
// ---------------------------------------------------------------------------
static GArgs gemm_default() {
  GArgs g;
  memset(&g, 0, sizeof(g));
  g.scale = 1.f; g.nb1 = 1; g.copies = 1;
  return g;
}
static void launch_gemm(const GArgs& g, int nb2, hipStream_t s) {
  dim3 grid(g.M / 16, g.N / 16, g.nb1 * nb2);
  k_gemm<<<grid, dim3(32), 0, s>>>(g);
}
static void launch_pack(const float* W, int K, int N, float scale, u16* out, hipStream_t s) {
  long long n = (long long)K * N;
  k_pack_wt<<<dim3((unsigned)((n + 255) / 256)), dim3(256), 0, s>>>(W, K, N, scale, out);
}

extern "C" void kernel_launch(void* const* d_in, const int* in_sizes, int n_in,
                              void* d_out, int out_size, void* d_ws, size_t ws_size,
                              hipStream_t stream) {
  (void)in_sizes; (void)n_in; (void)out_size; (void)ws_size;
  const float* rec_in  = (const float*)d_in[0];
  const float* lig_in  = (const float*)d_in[1];
  const float* pair_in = (const float*)d_in[2];
#define PW(i) ((const float*)d_in[i])

  float* outRec  = (float*)d_out;          // [256 x 64]
  float* outLig  = outRec + 16384;         // [384 x 64]
  float* outPair = outRec + 40960;         // [304 x 304 x 128]

  size_t off = 0;
  auto alloc = [&](size_t bytes) -> void* {
    size_t a = (off + 255) & ~(size_t)255;
    off = a + bytes;
    return (void*)((char*)d_ws + a);
  };
  const long long PAIRN = 304LL * 304 * 128;          // 11,829,248
  const long long LOG_H = (long long)NTOT * NPAD;     // 97,280
  const long long LOG_M = 8 * LOG_H;                  // 778,240

  // packed weights (bf16, [N x K])
  u16* pwRecQkv = (u16*)alloc(1536 * 64 * 2);
  u16* pwLigQkv = (u16*)alloc(1536 * 64 * 2);
  u16* pwBias4  = (u16*)alloc(32 * 128 * 2);
  u16* pwRecFin = (u16*)alloc(64 * 256 * 2);
  u16* pwLigFin = (u16*)alloc(64 * 256 * 2);
  u16* pwColKqv = (u16*)alloc(320 * 64 * 2);
  u16* pwColGate= (u16*)alloc(256 * 64 * 2);
  u16* pwColFin = (u16*)alloc(64 * 256 * 2);
  u16* pwRtr1   = (u16*)alloc(256 * 64 * 2);
  u16* pwRtr2   = (u16*)alloc(64 * 256 * 2);
  u16* pwLtr1   = (u16*)alloc(256 * 64 * 2);
  u16* pwLtr2   = (u16*)alloc(64 * 256 * 2);
  u16* pwPtr1   = (u16*)alloc(512 * 128 * 2);
  u16* pwPtr2   = (u16*)alloc(128 * 512 * 2);
  u16* pwOpmR   = (u16*)alloc(128 * 64 * 2);
  u16* pwOpmL   = (u16*)alloc(128 * 64 * 2);
  u16* pwoRR    = (u16*)alloc(4096 * 32 * 2);
  u16* pwoRL    = (u16*)alloc(4096 * 32 * 2);
  u16* pwoLR    = (u16*)alloc(4096 * 32 * 2);
  u16* pwoLL    = (u16*)alloc(4096 * 32 * 2);
  // activations / intermediates
  u16*   pairBF  = (u16*)alloc((size_t)PAIRN * 2);
  float* biasAll = (float*)alloc((size_t)304 * 304 * 32 * 4);
  u16*   recLn   = (u16*)alloc(256 * 64 * 2);
  u16*   ligLn   = (u16*)alloc(384 * 64 * 2);
  u16*   recQ6   = (u16*)alloc(256 * 1536 * 2);
  u16*   ligQ6   = (u16*)alloc(384 * 1536 * 2);
  u16*   recRq   = (u16*)alloc(8 * 256 * 32 * 2);
  u16*   recRk   = (u16*)alloc(8 * 256 * 32 * 2);
  u16*   recLq   = (u16*)alloc(8 * 256 * 32 * 2);
  u16*   recLk   = (u16*)alloc(8 * 256 * 32 * 2);
  u16*   recRvT  = (u16*)alloc(8 * 32 * 256 * 2);
  u16*   ligRk   = (u16*)alloc(8 * 384 * 32 * 2);
  u16*   ligRq   = (u16*)alloc(8 * 384 * 32 * 2);
  u16*   ligLq   = (u16*)alloc(8 * 384 * 32 * 2);
  u16*   ligLk   = (u16*)alloc(8 * 384 * 32 * 2);
  u16*   ligRvT  = (u16*)alloc(8 * 8 * 32 * 64 * 2);
  u16*   vcomb   = (u16*)alloc(64 * 32 * NPAD * 2);
  float* logits  = (float*)alloc((size_t)8 * LOG_M * 4);
  u16*   probs   = (u16*)alloc((size_t)8 * LOG_M * 2);
  u16*   recAtt  = (u16*)alloc(256 * 256 * 2);
  u16*   ligAtt  = (u16*)alloc(384 * 256 * 2);
  u16*   ligLn2  = (u16*)alloc(384 * 64 * 2);
  float* kqvF    = (float*)alloc(384 * NPAD * 4);
  float* gateF   = (float*)alloc(384 * 256 * 4);
  u16*   colAtt  = (u16*)alloc(384 * 256 * 2);
  u16*   recLn3  = (u16*)alloc(256 * 64 * 2);
  u16*   ligLn3  = (u16*)alloc(384 * 64 * 2);
  float* rpF     = (float*)alloc(256 * 128 * 4);
  float* lpF     = (float*)alloc(384 * 128 * 4);
  u16*   f_rri   = (u16*)alloc(256 * 32 * 2);
  u16*   f_rrj   = (u16*)alloc(256 * 32 * 2);
  u16*   f_rli   = (u16*)alloc(256 * 32 * 2);
  u16*   f_rlj   = (u16*)alloc(256 * 32 * 2);
  u16*   f_lmrj  = (u16*)alloc(48 * 32 * 2);
  u16*   f_lmri  = (u16*)alloc(48 * 32 * 2);
  u16*   f_llj   = (u16*)alloc(384 * 32 * 2);
  u16*   f_ali   = (u16*)alloc(48 * 256 * 2);
  u16*   U_rr    = (u16*)alloc((size_t)256 * 4096 * 2);
  u16*   U_rl    = (u16*)alloc((size_t)48 * 4096 * 2);
  u16*   U_lr    = (u16*)alloc((size_t)256 * 4096 * 2);
  u16*   U4raw   = (u16*)alloc((size_t)384 * 4096 * 2);
  u16*   U4      = (u16*)alloc((size_t)48 * 128 * 256 * 2);

  // ---------------- phase 0: weight packing ----------------
  launch_pack(PW(7),  64, 1536, 1.f, pwRecQkv, stream);
  launch_pack(PW(8),  64, 1536, 1.f, pwLigQkv, stream);
  k_pack_bias4<<<dim3(16), dim3(256), 0, stream>>>(PW(9), PW(10), PW(11), PW(12), pwBias4);
  launch_pack(PW(13), 256, 64, 1.f, pwRecFin, stream);
  launch_pack(PW(15), 256, 64, 1.f, pwLigFin, stream);
  launch_pack(PW(19), 64, 320, 1.f, pwColKqv, stream);
  launch_pack(PW(20), 64, 256, 1.f, pwColGate, stream);
  launch_pack(PW(22), 256, 64, 1.f, pwColFin, stream);
  launch_pack(PW(26), 64, 256, 1.f, pwRtr1, stream);
  launch_pack(PW(28), 256, 64, 1.f, pwRtr2, stream);
  launch_pack(PW(32), 64, 256, 1.f, pwLtr1, stream);
  launch_pack(PW(34), 256, 64, 1.f, pwLtr2, stream);
  launch_pack(PW(38), 128, 512, 1.f, pwPtr1, stream);
  launch_pack(PW(40), 512, 128, 1.f, pwPtr2, stream);
  launch_pack(PW(46), 64, 128, 1.f, pwOpmR, stream);
  launch_pack(PW(48), 64, 128, 1.f, pwOpmL, stream);
  k_pack_opm<<<dim3(512), dim3(256), 0, stream>>>(PW(50), 0, pwoRR);
  k_pack_opm<<<dim3(512), dim3(256), 0, stream>>>(PW(52), 0, pwoRL);
  k_pack_opm<<<dim3(512), dim3(256), 0, stream>>>(PW(54), 1, pwoLR);  // lr contracts x on U side
  k_pack_opm<<<dim3(512), dim3(256), 0, stream>>>(PW(56), 0, pwoLL);
  k_f2bf<<<dim3((unsigned)((PAIRN + 255) / 256)), dim3(256), 0, stream>>>(pair_in, PAIRN, pairBF);

  // ---------------- phase 1: row attention with pair bias ----------------
  k_ln<<<dim3(4), dim3(64), 0, stream>>>(rec_in, 256, 64, PW(3), PW(4), recLn);
  k_ln<<<dim3(6), dim3(64), 0, stream>>>(lig_in, 384, 64, PW(5), PW(6), ligLn);
  { GArgs g = gemm_default();                       // rec qkv6
    g.A = recLn; g.lda = 64; g.BT = pwRecQkv; g.ldb = 64;
    g.M = 256; g.N = 1536; g.K = 64; g.outH = recQ6; g.hRow = 1536;
    launch_gemm(g, 1, stream); }
  { GArgs g = gemm_default();                       // lig qkv6
    g.A = ligLn; g.lda = 64; g.BT = pwLigQkv; g.ldb = 64;
    g.M = 384; g.N = 1536; g.K = 64; g.outH = ligQ6; g.hRow = 1536;
    launch_gemm(g, 1, stream); }
  k_repack_rec<<<dim3(256), dim3(256), 0, stream>>>(recQ6, recRq, recRk, recLq, recLk, recRvT, vcomb);
  k_repack_lig<<<dim3(384), dim3(256), 0, stream>>>(ligQ6, ligRk, ligRq, ligLq, ligLk, ligRvT, vcomb);
  { GArgs g = gemm_default();                       // all 4 pair-bias projections at once
    g.A = pairBF; g.lda = 128; g.BT = pwBias4; g.ldb = 128;
    g.M = 304 * 304; g.N = 32; g.K = 128; g.outF = biasAll; g.oRow = 32;
    launch_gemm(g, 1, stream); }
  { GArgs g = gemm_default();                       // rr logits (broadcast over m)
    g.A = recRq; g.lda = 32; g.aB1 = 8192;
    g.BT = recRk; g.ldb = 32; g.bB1 = 8192;
    g.M = 256; g.N = 256; g.K = 32; g.nb1 = 8;
    g.outF = logits; g.oRow = NPAD; g.oB1 = LOG_H;
    g.copies = 8; g.copyStride = LOG_M;
    g.add2d = biasAll; g.addRow = NTOT * 32; g.addCol = 32; g.addB1 = 1;
    launch_gemm(g, 1, stream); }
  { GArgs g = gemm_default();                       // rl logits
    g.A = recLq; g.lda = 32; g.aB1 = 8192;
    g.BT = ligRk; g.ldb = 32; g.bB1 = 384 * 32; g.bB2 = 48 * 32;
    g.M = 256; g.N = 48; g.K = 32; g.nb1 = 8;
    g.outF = logits + 256; g.oRow = NPAD; g.oB1 = LOG_H; g.oB2 = LOG_M;
    g.add2d = biasAll + 256 * 32 + 8; g.addRow = NTOT * 32; g.addCol = 32; g.addB1 = 1;
    launch_gemm(g, 8, stream); }
  { GArgs g = gemm_default();                       // lr logits
    g.A = ligRq; g.lda = 32; g.aB1 = 384 * 32; g.aB2 = 48 * 32;
    g.BT = recLk; g.ldb = 32; g.bB1 = 8192;
    g.M = 48; g.N = 256; g.K = 32; g.nb1 = 8;
    g.outF = logits + 256 * NPAD; g.oRow = NPAD; g.oB1 = LOG_H; g.oB2 = LOG_M;
    g.add2d = biasAll + 256LL * NTOT * 32 + 16; g.addRow = NTOT * 32; g.addCol = 32; g.addB1 = 1;
    launch_gemm(g, 8, stream); }
  { GArgs g = gemm_default();                       // ll logits
    g.A = ligLq; g.lda = 32; g.aB1 = 384 * 32; g.aB2 = 48 * 32;
    g.BT = ligLk; g.ldb = 32; g.bB1 = 384 * 32; g.bB2 = 48 * 32;
    g.M = 48; g.N = 48; g.K = 32; g.nb1 = 8;
    g.outF = logits + 256 * NPAD + 256; g.oRow = NPAD; g.oB1 = LOG_H; g.oB2 = LOG_M;
    g.add2d = biasAll + (256LL * NTOT + 256) * 32 + 24; g.addRow = NTOT * 32; g.addCol = 32; g.addB1 = 1;
    launch_gemm(g, 8, stream); }
  k_softmax<<<dim3(8 * 8 * NTOT), dim3(32), 0, stream>>>(logits, probs);
  k_att_rec<<<dim3(16, 8), dim3(32), 0, stream>>>(probs, recRvT, ligRvT, recAtt);
  k_att_lig<<<dim3(3, 8, 8), dim3(32), 0, stream>>>(probs, vcomb, ligAtt);
  { GArgs g = gemm_default();                       // rec final proj + residual
    g.A = recAtt; g.lda = 256; g.BT = pwRecFin; g.ldb = 256;
    g.M = 256; g.N = 64; g.K = 256; g.bias = PW(14);
    g.resid = rec_in; g.resRow = 64; g.outF = outRec; g.oRow = 64;
    launch_gemm(g, 1, stream); }
  { GArgs g = gemm_default();                       // lig final proj + residual
    g.A = ligAtt; g.lda = 256; g.BT = pwLigFin; g.ldb = 256;
    g.M = 384; g.N = 64; g.K = 256; g.bias = PW(16);
    g.resid = lig_in; g.resRow = 64; g.outF = outLig; g.oRow = 64;
    launch_gemm(g, 1, stream); }

  // ---------------- phase 2: extra column global attention ----------------
  k_ln<<<dim3(6), dim3(64), 0, stream>>>(outLig, 384, 64, PW(17), PW(18), ligLn2);
  { GArgs g = gemm_default();                       // kqv
    g.A = ligLn2; g.lda = 64; g.BT = pwColKqv; g.ldb = 64;
    g.M = 384; g.N = 320; g.K = 64; g.outF = kqvF; g.oRow = NPAD;
    launch_gemm(g, 1, stream); }
  { GArgs g = gemm_default();                       // gate (no sigmoid per source)
    g.A = ligLn2; g.lda = 64; g.BT = pwColGate; g.ldb = 64;
    g.M = 384; g.N = 256; g.K = 64; g.bias = PW(21); g.outF = gateF; g.oRow = 256;
    launch_gemm(g, 1, stream); }
  k_colattn<<<dim3(6), dim3(64), 0, stream>>>(kqvF, gateF, colAtt);
  { GArgs g = gemm_default();                       // col final + residual (in-place)
    g.A = colAtt; g.lda = 256; g.BT = pwColFin; g.ldb = 256;
    g.M = 384; g.N = 64; g.K = 256; g.bias = PW(23);
    g.resid = outLig; g.resRow = 64; g.outF = outLig; g.oRow = 64;
    launch_gemm(g, 1, stream); }

  // ---------------- phase 3: rec / lig transitions (fused) ----------------
  k_transition<<<dim3(16), dim3(256), 0, stream>>>(outRec, 64, 256, PW(24), PW(25), pwRtr1, PW(27), pwRtr2, PW(29));
  k_transition<<<dim3(24), dim3(256), 0, stream>>>(outLig, 64, 256, PW(30), PW(31), pwLtr1, PW(33), pwLtr2, PW(35));

  // ---------------- phase 4: outer product mean -> pair ----------------
  hipMemcpyAsync(outPair, pair_in, (size_t)PAIRN * 4, hipMemcpyDeviceToDevice, stream);
  k_ln<<<dim3(4), dim3(64), 0, stream>>>(outRec, 256, 64, PW(42), PW(43), recLn3);
  k_ln<<<dim3(6), dim3(64), 0, stream>>>(outLig, 384, 64, PW(44), PW(45), ligLn3);
  { GArgs g = gemm_default();                       // rp
    g.A = recLn3; g.lda = 64; g.BT = pwOpmR; g.ldb = 64;
    g.M = 256; g.N = 128; g.K = 64; g.bias = PW(47); g.outF = rpF; g.oRow = 128;
    launch_gemm(g, 1, stream); }
  { GArgs g = gemm_default();                       // lp
    g.A = ligLn3; g.lda = 64; g.BT = pwOpmL; g.ldb = 64;
    g.M = 384; g.N = 128; g.K = 64; g.bias = PW(49); g.outF = lpF; g.oRow = 128;
    launch_gemm(g, 1, stream); }
  k_opm_factors<<<dim3(237), dim3(256), 0, stream>>>(rpF, lpF, f_rri, f_rrj, f_rli, f_rlj,
                                                     f_lmrj, f_lmri, f_llj, f_ali);
  { GArgs g = gemm_default();                       // U_rr[j] = r_rj · Wrr
    g.A = f_rrj; g.lda = 32; g.BT = pwoRR; g.ldb = 32;
    g.M = 256; g.N = 4096; g.K = 32; g.outH = U_rr; g.hRow = 4096;
    launch_gemm(g, 1, stream); }
  { GArgs g = gemm_default();                       // U_rl[ja] = mean_m(l_rj) · Wrl
    g.A = f_lmrj; g.lda = 32; g.BT = pwoRL; g.ldb = 32;
    g.M = 48; g.N = 4096; g.K = 32; g.outH = U_rl; g.hRow = 4096;
    launch_gemm(g, 1, stream); }
  { GArgs g = gemm_default();                       // U_lr[i] = r_lj · Wlr (x-contract)
    g.A = f_rlj; g.lda = 32; g.BT = pwoLR; g.ldb = 32;
    g.M = 256; g.N = 4096; g.K = 32; g.outH = U_lr; g.hRow = 4096;
    launch_gemm(g, 1, stream); }
  { GArgs g = gemm_default();                       // U4raw[m,jb] = l_lj · Wll
    g.A = f_llj; g.lda = 32; g.BT = pwoLL; g.ldb = 32;
    g.M = 384; g.N = 4096; g.K = 32; g.outH = U4raw; g.hRow = 4096;
    launch_gemm(g, 1, stream); }
  k_repack_u4<<<dim3(6144), dim3(256), 0, stream>>>(U4raw, U4);
  { GArgs g = gemm_default();                       // pair[rr] += r_ri · U_rr[j]
    g.A = f_rri; g.lda = 32; g.BT = U_rr; g.ldb = 32; g.bB1 = 4096;
    g.M = 256; g.N = 128; g.K = 32; g.nb1 = 256;
    g.outF = outPair; g.oRow = (long long)NTOT * 128; g.oB1 = 128;
    g.accumulate = 1; g.bias = PW(51);
    launch_gemm(g, 1, stream); }
  { GArgs g = gemm_default();                       // pair[rl] += r_li · U_rl[ja]
    g.A = f_rli; g.lda = 32; g.BT = U_rl; g.ldb = 32; g.bB1 = 4096;
    g.M = 256; g.N = 128; g.K = 32; g.nb1 = 48;
    g.outF = outPair + 256LL * 128; g.oRow = (long long)NTOT * 128; g.oB1 = 128;
    g.accumulate = 1; g.bias = PW(53);
    launch_gemm(g, 1, stream); }
  { GArgs g = gemm_default();                       // pair[lr] += lm_ri · U_lr[i]
    g.A = f_lmri; g.lda = 32; g.BT = U_lr; g.ldb = 32; g.bB1 = 4096;
    g.M = 48; g.N = 128; g.K = 32; g.nb1 = 256;
    g.outF = outPair + 256LL * NTOT * 128; g.oRow = (long long)NTOT * 128; g.oB1 = 128;
    g.accumulate = 1; g.bias = PW(55);
    launch_gemm(g, 1, stream); }
  { GArgs g = gemm_default();                       // pair[ll] += (l_li/M) · U4[jb]
    g.A = f_ali; g.lda = 256; g.BT = U4; g.ldb = 256; g.bB1 = 128 * 256;
    g.M = 48; g.N = 128; g.K = 256; g.nb1 = 48;
    g.outF = outPair + (256LL * NTOT + 256) * 128; g.oRow = (long long)NTOT * 128; g.oB1 = 128;
    g.accumulate = 1; g.bias = PW(57);
    launch_gemm(g, 1, stream); }

  // ---------------- phase 5: pair transition (fused, 5776 blocks) ----------------
  k_transition<<<dim3(5776), dim3(256), 0, stream>>>(outPair, 128, 512, PW(36), PW(37),
                                                     pwPtr1, PW(39), pwPtr2, PW(41));
#undef PW
}
